// TinyLlamaBlock_53601191854759
// MI455X (gfx1250) — matrix operand order repaired and checked by reading, not compiled, not run on hardware
//
#include <hip/hip_runtime.h>

// ---------------------------------------------------------------------------
// TinyLlama block on gfx1250 (MI455X): bf16 WMMA GEMMs + flash attention.
// B=2, S=2048, E=2048, H=16, D=128.  Dense math on v_wmma_f32_16x16x32_bf16,
// GEMM staging on global_load_async_to_lds_b128 (ASYNCcnt double buffering).
// ---------------------------------------------------------------------------

typedef __attribute__((ext_vector_type(16))) __bf16 v16bf;
typedef __attribute__((ext_vector_type(8)))  float  v8f;

#define Bb 2
#define Ss 2048
#define Ee 2048
#define Hh 16
#define Dd 128
#define Mm 4096   // B*S
#define LDS_STRIDE 40   // bf16 elems per row (32 data + 8 pad), 80B => 16B aligned rows

union FragU { v16bf v; unsigned u[8]; };

__device__ inline v8f wmma_bf16(v16bf a, v16bf b, v8f c) {
  return __builtin_amdgcn_wmma_f32_16x16x32_bf16(false, a, false, b, (short)0, c,
                                                 false, false);
}

// Async DMA: global -> LDS, 16B per lane, tracked by ASYNCcnt (no VGPR data).
// VDST carries the LDS byte address; flat-aperture rule: LDS offset = addr[31:0].
__device__ inline void async_load_b128(unsigned lds_addr, const void* gaddr) {
  asm volatile("global_load_async_to_lds_b128 %0, %1, off"
               :: "v"(lds_addr), "v"(gaddr) : "memory");
}
__device__ inline unsigned lds_off(const void* p) {
  return (unsigned)(size_t)p;   // addr[31:0] == LDS offset per aperture spec
}

// A-matrix 16x32 bf16 fragment (ISA 7.12.2): lane m (both halves) holds row m;
// VGPR v holds K pair k0 = (v>>2)*16 + half*8 + (v&3)*2.
__device__ inline v16bf load_fragA(const __bf16* __restrict__ base, int stride) {
  const int lane = threadIdx.x & 31;
  const int half = lane >> 4, mr = lane & 15;
  FragU f;
  const __bf16* row = base + (size_t)mr * stride;
#pragma unroll
  for (int v = 0; v < 8; ++v) {
    const int k0 = ((v >> 2) << 4) + (half << 3) + ((v & 3) << 1);
    f.u[v] = *(const unsigned*)(row + k0);
  }
  return f.v;
}

// B-matrix 32x16 bf16 fragment from K-contiguous [n][k] storage: lane holds
// column n = lane&15; VGPR v holds K pair k0 = (v>>2)*16 + half*8 + (v&3)*2.
__device__ inline v16bf load_fragB(const __bf16* __restrict__ base, int stride) {
  const int lane = threadIdx.x & 31;
  const int half = lane >> 4, nc = lane & 15;
  FragU f;
  const __bf16* row = base + (size_t)nc * stride;
#pragma unroll
  for (int v = 0; v < 8; ++v) {
    const int k0 = ((v >> 2) << 4) + (half << 3) + ((v & 3) << 1);
    f.u[v] = *(const unsigned*)(row + k0);
  }
  return f.v;
}

// ---------------------------------------------------------------------------
// Weight convert + transpose: W[k][n] f32 -> WT[n][k] bf16  (K = N = 2048)
// ---------------------------------------------------------------------------
__global__ __launch_bounds__(256) void convT_kernel(const float* __restrict__ W,
                                                    __bf16* __restrict__ WT) {
  const int idx = blockIdx.x * 256 + threadIdx.x;       // 0 .. 4M-1
  const int k = idx >> 11;
  const int n = idx & 2047;
  WT[(size_t)n * Ee + k] = (__bf16)W[idx];
}

// ---------------------------------------------------------------------------
// RMSNorm: one row (E=2048) per block, f32 in -> bf16 out
// ---------------------------------------------------------------------------
__global__ __launch_bounds__(256) void rmsnorm_kernel(const float* __restrict__ x,
                                                      const float* __restrict__ scale,
                                                      __bf16* __restrict__ out) {
  __shared__ float red[256];
  const int row = blockIdx.x;
  const float* xr = x + (size_t)row * Ee;
  float s = 0.f;
#pragma unroll
  for (int it = 0; it < 2; ++it) {
    const int i = threadIdx.x + it * 256;               // float4 index, 512 total
    float4 v = ((const float4*)xr)[i];
    s += v.x * v.x + v.y * v.y + v.z * v.z + v.w * v.w;
  }
  red[threadIdx.x] = s;
  __syncthreads();
  for (int o = 128; o > 0; o >>= 1) {
    if (threadIdx.x < o) red[threadIdx.x] += red[threadIdx.x + o];
    __syncthreads();
  }
  const float inv = rsqrtf(red[0] * (1.0f / Ee) + 1e-5f);
  __bf16* orow = out + (size_t)row * Ee;
#pragma unroll
  for (int it = 0; it < 2; ++it) {
    const int i = threadIdx.x + it * 256;
    float4 v  = ((const float4*)xr)[i];
    float4 sc = ((const float4*)scale)[i];
    orow[i * 4 + 0] = (__bf16)(v.x * inv * sc.x);
    orow[i * 4 + 1] = (__bf16)(v.y * inv * sc.y);
    orow[i * 4 + 2] = (__bf16)(v.z * inv * sc.z);
    orow[i * 4 + 3] = (__bf16)(v.w * inv * sc.w);
  }
}

// ---------------------------------------------------------------------------
// GEMM: C[M,N] = A[M,K](bf16) * BT[N,K](bf16), f32 accum via WMMA.
// Block 256 thr = 8 waves, tile 128x128, BK=32. Wave: 64x32 = 8 WMMA tiles.
// Double-buffered LDS staging via global_load_async_to_lds_b128 (ASYNCcnt).
// mode 0: store bf16 C[m*N+n]
// mode 1: store bf16 V-transposed -> [B,H,D,S]
// mode 2: store f32 C + resid
// mode 3: store f32 C + resid + bias[n]
// ---------------------------------------------------------------------------
__global__ __launch_bounds__(256) void gemm_bf16_kernel(
    const __bf16* __restrict__ A, const __bf16* __restrict__ BT,
    float* __restrict__ outF, __bf16* __restrict__ outB,
    const float* __restrict__ resid, const float* __restrict__ bias,
    int N, int K, int mode) {
  __shared__ __bf16 ldsA[2][128 * LDS_STRIDE];
  __shared__ __bf16 ldsB[2][128 * LDS_STRIDE];
  const int m0 = blockIdx.y * 128;
  const int n0 = blockIdx.x * 128;
  const int tid = threadIdx.x;
  const int wave = tid >> 5;
  const int wr = wave >> 2;   // 0..1 -> 64-row strip
  const int wc = wave & 3;    // 0..3 -> 32-col strip

  const v8f vzero = {0.f, 0.f, 0.f, 0.f, 0.f, 0.f, 0.f, 0.f};
  v8f acc[4][2];
#pragma unroll
  for (int i = 0; i < 4; ++i)
#pragma unroll
    for (int j = 0; j < 2; ++j) acc[i][j] = vzero;

  // issue one K-slab (128x32 of A and of BT) as 4 async b128 DMAs per thread
  auto issue_stage = [&](int stage, int kb) {
#pragma unroll
    for (int it = 0; it < 2; ++it) {
      const int c = tid + it * 256;          // 512 16B chunks per matrix
      const int row = c >> 2, seg = c & 3;
      async_load_b128(lds_off(&ldsA[stage][row * LDS_STRIDE + seg * 8]),
                      A + (size_t)(m0 + row) * K + kb + seg * 8);
      async_load_b128(lds_off(&ldsB[stage][row * LDS_STRIDE + seg * 8]),
                      BT + (size_t)(n0 + row) * K + kb + seg * 8);
    }
  };

  const int T = K >> 5;        // K/32 slabs
  issue_stage(0, 0);
  for (int i = 0; i < T; ++i) {
    const int cur = i & 1;
    const bool hasNext = (i + 1) < T;
    if (hasNext) {
      issue_stage(cur ^ 1, (i + 1) << 5);
      // 8 async ops in flight; in-order completion => <=4 releases stage `cur`
      asm volatile("s_wait_asynccnt 0x4" ::: "memory");
    } else {
      asm volatile("s_wait_asynccnt 0x0" ::: "memory");
    }
    __syncthreads();

    v16bf af[4], bf[2];
#pragma unroll
    for (int ii = 0; ii < 4; ++ii)
      af[ii] = load_fragA(&ldsA[cur][(wr * 64 + ii * 16) * LDS_STRIDE], LDS_STRIDE);
#pragma unroll
    for (int j = 0; j < 2; ++j)
      bf[j] = load_fragB(&ldsB[cur][(wc * 32 + j * 16) * LDS_STRIDE], LDS_STRIDE);
#pragma unroll
    for (int ii = 0; ii < 4; ++ii)
#pragma unroll
      for (int j = 0; j < 2; ++j) acc[ii][j] = wmma_bf16(af[ii], bf[j], acc[ii][j]);
    __syncthreads();   // reads done before stage `cur` is re-filled at i+2
  }

  // epilogue: C layout — VGPR v holds row half*8+v, lane&15 is column
  const int lane = tid & 31, half = lane >> 4, col = lane & 15;
#pragma unroll
  for (int i = 0; i < 4; ++i)
#pragma unroll
    for (int j = 0; j < 2; ++j)
#pragma unroll
      for (int v = 0; v < 8; ++v) {
        const int m = m0 + wr * 64 + i * 16 + half * 8 + v;
        const int n = n0 + wc * 32 + j * 16 + col;
        const float val = acc[i][j][v];
        if (mode == 0) {
          outB[(size_t)m * N + n] = (__bf16)val;
        } else if (mode == 1) {
          const int b = m >> 11, s = m & 2047;        // S = 2048
          const int h = n >> 7, d = n & 127;          // D = 128
          outB[(((size_t)(b * Hh + h) * Dd + d) << 11) + s] = (__bf16)val;
        } else {
          float r = resid[(size_t)m * N + n];
          if (mode == 3) r += bias[n];
          outF[(size_t)m * N + n] = val + r;
        }
      }
}

// ---------------------------------------------------------------------------
// Flash attention: one wave per (b, h, 16-query tile); key blocks of 32.
// Q:[B,S,H,D] bf16, K:[B,S,H,D] bf16, Vt:[B,H,D,S] bf16, ctx:[B,S,H*D] bf16.
// ---------------------------------------------------------------------------
__global__ __launch_bounds__(256) void attn_kernel(const __bf16* __restrict__ Q,
                                                   const __bf16* __restrict__ Km,
                                                   const __bf16* __restrict__ Vt,
                                                   __bf16* __restrict__ ctx) {
  __shared__ __bf16 ldsP[8][16 * LDS_STRIDE];
  const int tid = threadIdx.x;
  const int wave = tid >> 5;
  const int lane = tid & 31, half = lane >> 4, lcol = lane & 15;

  const int w  = blockIdx.x * 8 + wave;  // 0..4095
  const int b  = w >> 11;
  const int h  = (w >> 7) & 15;
  const int qt = w & 127;
  const int q0 = qt * 16;

  const __bf16* qptr = Q  + ((size_t)(b * Ss + q0)) * Ee + h * Dd;
  const __bf16* kptr = Km + ((size_t)b * Ss) * Ee + h * Dd;
  const __bf16* vptr = Vt + ((size_t)(b * Hh + h)) * Dd * Ss;

  v16bf qf[4];
#pragma unroll
  for (int dc = 0; dc < 4; ++dc) qf[dc] = load_fragA(qptr + dc * 32, Ee);

  const v8f vzero = {0.f, 0.f, 0.f, 0.f, 0.f, 0.f, 0.f, 0.f};
  v8f o[8];
#pragma unroll
  for (int d8 = 0; d8 < 8; ++d8) o[d8] = vzero;
  float mrow[8], lrow[8];
#pragma unroll
  for (int v = 0; v < 8; ++v) { mrow[v] = -3.0e38f; lrow[v] = 0.f; }

  const float scale = 0.08838834764831845f;  // 1/sqrt(128)
  const int nkb = (q0 + 47) >> 5;            // key blocks of 32 up to diagonal

  for (int kb = 0; kb < nkb; ++kb) {
    const int ks = kb * 32;
    if (kb + 1 < nkb) {  // hint next key/value block toward L2 (global_prefetch_b8)
      __builtin_prefetch(kptr + (size_t)(ks + 32 + lane) * Ee, 0, 1);
      __builtin_prefetch(vptr + (size_t)(lane * 4) * Ss + ks + 32, 0, 1);
    }
    // ---- S = Q * K^T (two 16x16 tiles, K accumulated over D in 4 steps) ----
    v8f sc[2];
#pragma unroll
    for (int nt = 0; nt < 2; ++nt) {
      v8f s = vzero;
#pragma unroll
      for (int dc = 0; dc < 4; ++dc) {
        v16bf kf = load_fragB(kptr + (size_t)(ks + nt * 16) * Ee + dc * 32, Ee);
        s = wmma_bf16(qf[dc], kf, s);
      }
      sc[nt] = s;
    }
    // ---- scale + causal mask ----
    const bool needmask = (ks + 31) > q0;
#pragma unroll
    for (int nt = 0; nt < 2; ++nt)
#pragma unroll
      for (int v = 0; v < 8; ++v) {
        float sv = sc[nt][v] * scale;
        if (needmask) {
          const int cg = ks + nt * 16 + lcol;
          const int rg = q0 + half * 8 + v;
          if (cg > rg) sv = -3.0e38f;
        }
        sc[nt][v] = sv;
      }
    // ---- online softmax: rows live across the 16 lanes of each half ----
#pragma unroll
    for (int v = 0; v < 8; ++v) {
      float tm = fmaxf(sc[0][v], sc[1][v]);
      for (int off = 8; off > 0; off >>= 1) tm = fmaxf(tm, __shfl_xor(tm, off, 16));
      const float mn = fmaxf(mrow[v], tm);
      const float fs = __expf(mrow[v] - mn);
      const float p0 = __expf(sc[0][v] - mn);
      const float p1 = __expf(sc[1][v] - mn);
      sc[0][v] = p0; sc[1][v] = p1;
      float rs = p0 + p1;
      for (int off = 8; off > 0; off >>= 1) rs += __shfl_xor(rs, off, 16);
      lrow[v] = lrow[v] * fs + rs;
      mrow[v] = mn;
#pragma unroll
      for (int d8 = 0; d8 < 8; ++d8) o[d8][v] *= fs;
    }
    // ---- transpose P (C layout -> A layout) through per-wave LDS ----
    __bf16* pb = ldsP[wave];
#pragma unroll
    for (int nt = 0; nt < 2; ++nt)
#pragma unroll
      for (int v = 0; v < 8; ++v)
        pb[(half * 8 + v) * LDS_STRIDE + nt * 16 + lcol] = (__bf16)sc[nt][v];
    asm volatile("s_wait_dscnt 0" ::: "memory");  // cross-lane LDS dependency
    v16bf pf = load_fragA(pb, LDS_STRIDE);
    // ---- O += P * V (Vt gives contiguous key pairs per d row) ----
#pragma unroll
    for (int d8 = 0; d8 < 8; ++d8) {
      v16bf vf = load_fragB(vptr + (size_t)(d8 * 16) * Ss + ks, Ss);
      o[d8] = wmma_bf16(pf, vf, o[d8]);
    }
  }

  // ---- normalize and store ctx as bf16 [B,S,H*D] ----
#pragma unroll
  for (int v = 0; v < 8; ++v) {
    const float inv = 1.0f / lrow[v];
    const int rg = q0 + half * 8 + v;
    const size_t rowoff = ((size_t)(b * Ss + rg)) * Ee + h * Dd;
#pragma unroll
    for (int d8 = 0; d8 < 8; ++d8)
      ctx[rowoff + d8 * 16 + lcol] = (__bf16)(o[d8][v] * inv);
  }
}

// ---------------------------------------------------------------------------
extern "C" void kernel_launch(void* const* d_in, const int* in_sizes, int n_in,
                              void* d_out, int out_size, void* d_ws, size_t ws_size,
                              hipStream_t stream) {
  const float* x    = (const float*)d_in[0];
  const float* ln1  = (const float*)d_in[1];
  const float* Wq   = (const float*)d_in[2];
  const float* Wk   = (const float*)d_in[3];
  const float* Wv   = (const float*)d_in[4];
  const float* Wo   = (const float*)d_in[5];
  const float* ln2  = (const float*)d_in[6];
  const float* Wmlp = (const float*)d_in[7];
  const float* bmlp = (const float*)d_in[8];
  float* out = (float*)d_out;

  char* ws = (char*)d_ws;
  const size_t MB = 1024 * 1024;
  __bf16* xn   = (__bf16*)(ws + 0);          // 16 MB  (also reused as ctx)
  __bf16* WqT  = (__bf16*)(ws + 16 * MB);    //  8 MB each
  __bf16* WkT  = (__bf16*)(ws + 24 * MB);
  __bf16* WvT  = (__bf16*)(ws + 32 * MB);
  __bf16* WoT  = (__bf16*)(ws + 40 * MB);
  __bf16* WmT  = (__bf16*)(ws + 48 * MB);
  __bf16* Qb   = (__bf16*)(ws + 56 * MB);    // 16 MB
  __bf16* Kb   = (__bf16*)(ws + 72 * MB);    // 16 MB
  __bf16* Vt   = (__bf16*)(ws + 88 * MB);    // 16 MB  (total 104 MB)
  __bf16* ctx  = xn;                          // xn dead after QKV GEMMs
  float*  x1   = (float*)(ws + 56 * MB);      // overlays Qb+Kb (dead after attn)
  __bf16* x1n  = (__bf16*)(ws + 88 * MB);     // overlays Vt   (dead after attn)

  const dim3 gConv(Ee * Ee / 256), bConv(256);
  convT_kernel<<<gConv, bConv, 0, stream>>>(Wq,   WqT);
  convT_kernel<<<gConv, bConv, 0, stream>>>(Wk,   WkT);
  convT_kernel<<<gConv, bConv, 0, stream>>>(Wv,   WvT);
  convT_kernel<<<gConv, bConv, 0, stream>>>(Wo,   WoT);
  convT_kernel<<<gConv, bConv, 0, stream>>>(Wmlp, WmT);

  rmsnorm_kernel<<<Mm, 256, 0, stream>>>(x, ln1, xn);

  const dim3 gGemm(Ee / 128, Mm / 128), bGemm(256);
  gemm_bf16_kernel<<<gGemm, bGemm, 0, stream>>>(xn, WqT, nullptr, Qb, nullptr, nullptr, Ee, Ee, 0);
  gemm_bf16_kernel<<<gGemm, bGemm, 0, stream>>>(xn, WkT, nullptr, Kb, nullptr, nullptr, Ee, Ee, 0);
  gemm_bf16_kernel<<<gGemm, bGemm, 0, stream>>>(xn, WvT, nullptr, Vt, nullptr, nullptr, Ee, Ee, 1);

  attn_kernel<<<Bb * Hh * (Ss / 16) / 8, 256, 0, stream>>>(Qb, Kb, Vt, ctx);

  gemm_bf16_kernel<<<gGemm, bGemm, 0, stream>>>(ctx, WoT, x1, nullptr, x, nullptr, Ee, Ee, 2);
  rmsnorm_kernel<<<Mm, 256, 0, stream>>>(x1, ln2, x1n);
  gemm_bf16_kernel<<<gGemm, bGemm, 0, stream>>>(x1n, WmT, out, nullptr, x1, bmlp, Ee, Ee, 3);
}
